// FastSelfAttention_86663850099257
// MI455X (gfx1250) — matrix operand order repaired and checked by reading, not compiled, not run on hardware
//
#include <hip/hip_runtime.h>
#include <math.h>

// ---------------------------------------------------------------------------
// FAVOR+ linear attention for MI455X (gfx1250, wave32)
// WMMA f32_16x16x32_f16 compute + Tensor Data Mover (TDM) LDS staging,
// double-buffered on TENSORcnt, K-slab = 64 (16 WMMAs per sync window).
// ---------------------------------------------------------------------------

typedef __attribute__((ext_vector_type(16))) _Float16 v16h;
typedef __attribute__((ext_vector_type(8)))  float    v8f;
typedef __attribute__((ext_vector_type(4)))  unsigned u32x4;
typedef __attribute__((ext_vector_type(8)))  unsigned u32x8;

struct __attribute__((aligned(16))) U4 { unsigned x, y, z, w; };

__device__ __forceinline__ v8f wmma32(v16h a, v16h b, v8f c) {
  return __builtin_amdgcn_wmma_f32_16x16x32_f16(
      false, a, false, b, (short)0, c, false, false);
}

// One TDM 2D tile load: tile_rows x tile_cols halves, row stride stride_elems,
// into contiguous LDS at lds_dst. Wave-uniform; EXEC ignored by TDM.
__device__ __forceinline__ void tdm_load_2d(const void* lds_dst, const void* gsrc,
                                            unsigned tile_rows, unsigned tile_cols,
                                            unsigned stride_elems) {
  unsigned long long ga = (unsigned long long)(size_t)gsrc;
  u32x4 g0;
  g0[0] = 1u;                                   // count=1 (valid user D#)
  g0[1] = (unsigned)(size_t)lds_dst;            // lds_addr (bytes)
  g0[2] = (unsigned)ga;                         // global_addr[31:0]
  g0[3] = ((unsigned)(ga >> 32) & 0x01FFFFFFu) | (2u << 30);  // addr[56:32], type=2
  u32x8 g1;
  g1[0] = 1u << 16;                             // wg_mask=0, data_size=1 (2B)
  g1[1] = (stride_elems & 0xFFFFu) << 16;       // tensor_dim0[15:0]
  g1[2] = (stride_elems >> 16) | ((tile_rows & 0xFFFFu) << 16);  // dim0 hi, dim1 lo
  g1[3] = (tile_rows >> 16) | (tile_cols << 16);// dim1 hi, tile_dim0
  g1[4] = tile_rows & 0xFFFFu;                  // tile_dim1 = rows, tile_dim2 = 0
  g1[5] = stride_elems;                         // tensor_dim0_stride[31:0]
  g1[6] = 0;
  g1[7] = 0;
  asm volatile("tensor_load_to_lds %0, %1" :: "s"(g0), "s"(g1) : "memory");
}

// A fragment from LDS tile [rows][ldk] halves (row-major).
__device__ __forceinline__ v16h frag_a(const _Float16* base, int row0, int k0,
                                       int lane, int ldk) {
  const _Float16* p = base + (row0 + (lane & 15)) * ldk + k0 + ((lane >> 4) << 3);
  v16h f;
  *(U4*)&f       = *(const U4*)p;
  *((U4*)&f + 1) = *(const U4*)(p + 16);
  return f;
}
// B fragment from LDS tile [cols][ldk] halves ([N][K]).
__device__ __forceinline__ v16h frag_b(const _Float16* base, int col0, int k0,
                                       int lane, int ldk) {
  const _Float16* p = base + (col0 + (lane & 15)) * ldk + k0 + ((lane >> 4) << 3);
  v16h f;
  *(U4*)&f       = *(const U4*)p;
  *((U4*)&f + 1) = *(const U4*)(p + 16);
  return f;
}
__device__ __forceinline__ void cp16(_Float16* dst, const _Float16* src) {
  *(U4*)dst       = *(const U4*)src;
  *(U4*)(dst + 8) = *(const U4*)(src + 8);
}

// ------------------------------ converters ---------------------------------

__global__ void k_cvt_f16(const float* __restrict__ in, _Float16* __restrict__ out, int n) {
  int i = blockIdx.x * blockDim.x + threadIdx.x;
  if (i < n) out[i] = (_Float16)in[i];
}

__global__ void k_pack_wqkvT(const float* __restrict__ Wq, const float* __restrict__ Wk,
                             const float* __restrict__ Wv, _Float16* __restrict__ out) {
  int i = blockIdx.x * blockDim.x + threadIdx.x;
  if (i >= 3072 * 1024) return;
  int n = i >> 10, k = i & 1023;
  int sel = n >> 10, nn = n & 1023;
  const float* W = (sel == 0) ? Wq : (sel == 1) ? Wk : Wv;
  out[i] = (_Float16)W[k * 1024 + nn];
}
__global__ void k_pack_woT(const float* __restrict__ Wo, _Float16* __restrict__ out) {
  int i = blockIdx.x * blockDim.x + threadIdx.x;
  if (i >= 1024 * 1024) return;
  int n = i >> 10, k = i & 1023;
  out[i] = (_Float16)Wo[k * 1024 + n];
}
__global__ void k_pack_rffwT(const float* __restrict__ W, _Float16* __restrict__ out) {
  int i = blockIdx.x * blockDim.x + threadIdx.x;
  if (i >= 256 * 64) return;
  int n = i >> 6, k = i & 63;
  out[i] = (_Float16)W[k * 256 + n];
}

// ------------------------ GEMM 1: fused QKV projection ----------------------
__global__ __launch_bounds__(256) void k_gemm_qkv(
    const _Float16* __restrict__ A, const _Float16* __restrict__ BwT,
    const float* __restrict__ bq, const float* __restrict__ bk, const float* __restrict__ bv,
    _Float16* __restrict__ qh, _Float16* __restrict__ kh, _Float16* __restrict__ vhT) {
  __shared__ __align__(16) _Float16 As[2][128 * 64];
  __shared__ __align__(16) _Float16 Bs[2][128 * 64];
  const int t = threadIdx.x, lane = t & 31, wave = t >> 5;
  const int wr = wave >> 2, wc = wave & 3;            // 2x4 waves, wave tile 64x32
  const int m0 = blockIdx.y * 128, n0 = blockIdx.x * 128;
  const bool issuer = (wave == 0);
  const _Float16* pA = A + (size_t)m0 * 1024;
  const _Float16* pB = BwT + (size_t)n0 * 1024;

  if (issuer) {
    tdm_load_2d(As[0], pA, 128, 64, 1024);
    tdm_load_2d(Bs[0], pB, 128, 64, 1024);
  }
  v8f acc[4][2] = {};
  for (int kt = 0; kt < 1024; kt += 64) {
    const int cur = (kt >> 6) & 1;
    if (kt + 64 < 1024) {
      if (issuer) {
        tdm_load_2d(As[1 - cur], pA + kt + 64, 128, 64, 1024);
        tdm_load_2d(Bs[1 - cur], pB + kt + 64, 128, 64, 1024);
      }
      __builtin_amdgcn_s_wait_tensorcnt(2);
    } else {
      __builtin_amdgcn_s_wait_tensorcnt(0);
    }
    __syncthreads();
#pragma unroll
    for (int k2 = 0; k2 < 64; k2 += 32) {
      v16h af[4], bf[2];
#pragma unroll
      for (int i = 0; i < 4; ++i) af[i] = frag_a(As[cur], wr * 64 + i * 16, k2, lane, 64);
#pragma unroll
      for (int j = 0; j < 2; ++j) bf[j] = frag_b(Bs[cur], wc * 32 + j * 16, k2, lane, 64);
#pragma unroll
      for (int i = 0; i < 4; ++i)
#pragma unroll
        for (int j = 0; j < 2; ++j) acc[i][j] = wmma32(af[i], bf[j], acc[i][j]);
    }
    __syncthreads();
  }
#pragma unroll
  for (int i = 0; i < 4; ++i) {
    int mb = m0 + wr * 64 + i * 16 + ((lane >> 4) << 3);
#pragma unroll
    for (int j = 0; j < 2; ++j) {
      int n = n0 + wc * 32 + j * 16 + (lane & 15);
      int sel = n >> 10, nn = n & 1023;
      int hh = nn >> 6, d = nn & 63;
      float bias = ((sel == 0) ? bq : (sel == 1) ? bk : bv)[nn];
#pragma unroll
      for (int r = 0; r < 8; ++r) {
        int m = mb + r;
        int b = m >> 12, s = m & 4095;
        int bh = b * 16 + hh;
        float v = acc[i][j][r] + bias;
        if (sel == 2) {
          vhT[((size_t)bh * 64 + d) * 4096 + s] = (_Float16)v;
        } else {
          _Float16* dst = (sel == 0) ? qh : kh;
          dst[((size_t)bh * 4096 + s) * 64 + d] = (_Float16)(v * 0.125f);  // NORM
        }
      }
    }
  }
}

// --------------------- GEMM 2: random-Fourier features ----------------------
// K = 64 -> single TDM pair, no K-loop.
__global__ __launch_bounds__(256) void k_gemm_rff(
    const _Float16* __restrict__ A, const _Float16* __restrict__ BwT,
    const float* __restrict__ rffb, _Float16* __restrict__ out, int transposed) {
  __shared__ __align__(16) _Float16 As[128 * 64];
  __shared__ __align__(16) _Float16 Bs[128 * 64];
  const int t = threadIdx.x, lane = t & 31, wave = t >> 5;
  const int wr = wave >> 2, wc = wave & 3;
  const int m0 = blockIdx.y * 128, n0 = blockIdx.x * 128;

  if (wave == 0) {
    tdm_load_2d(As, A + (size_t)m0 * 64, 128, 64, 64);
    tdm_load_2d(Bs, BwT + (size_t)n0 * 64, 128, 64, 64);
  }
  __builtin_amdgcn_s_wait_tensorcnt(0);
  __syncthreads();

  v8f acc[4][2] = {};
#pragma unroll
  for (int k2 = 0; k2 < 64; k2 += 32) {
    v16h af[4], bf[2];
#pragma unroll
    for (int i = 0; i < 4; ++i) af[i] = frag_a(As, wr * 64 + i * 16, k2, lane, 64);
#pragma unroll
    for (int j = 0; j < 2; ++j) bf[j] = frag_b(Bs, wc * 32 + j * 16, k2, lane, 64);
#pragma unroll
    for (int i = 0; i < 4; ++i)
#pragma unroll
      for (int j = 0; j < 2; ++j) acc[i][j] = wmma32(af[i], bf[j], acc[i][j]);
  }
  const float sc2 = 0.0883883476f;  // sqrt(2/256)
#pragma unroll
  for (int i = 0; i < 4; ++i) {
    int mb = m0 + wr * 64 + i * 16 + ((lane >> 4) << 3);
#pragma unroll
    for (int j = 0; j < 2; ++j) {
      int n = n0 + wc * 32 + j * 16 + (lane & 15);
      float b = rffb[n];
#pragma unroll
      for (int r = 0; r < 8; ++r) {
        int m = mb + r;
        _Float16 v = (_Float16)(cosf(acc[i][j][r] + b) * sc2);
        if (transposed)
          out[((size_t)(m >> 12) * 256 + n) * 4096 + (m & 4095)] = v;
        else
          out[(size_t)m * 256 + n] = v;
      }
    }
  }
}

// ------------------------------- k_sum --------------------------------------
__global__ __launch_bounds__(256) void k_ksum(const _Float16* __restrict__ kpT,
                                              float* __restrict__ ksum) {
  int bh = blockIdx.x, f = threadIdx.x;
  const _Float16* p = kpT + ((size_t)bh * 256 + f) * 4096;
  float s = 0.f;
  for (int i = 0; i < 4096; i += 8) {
    __align__(16) _Float16 h[8];
    *(U4*)h = *(const U4*)(p + i);
#pragma unroll
    for (int j = 0; j < 8; ++j) s += (float)h[j];
  }
  ksum[bh * 256 + f] = s;
}

// ------------------- GEMM 3: kv = k'^T v  (per head) ------------------------
__global__ __launch_bounds__(256) void k_gemm_kv(
    const _Float16* __restrict__ KpT, const _Float16* __restrict__ VhT,
    _Float16* __restrict__ KvT) {
  __shared__ __align__(16) _Float16 As[2][128 * 64];  // [f][s]
  __shared__ __align__(16) _Float16 Bs[2][64 * 64];   // [d][s]
  const int t = threadIdx.x, lane = t & 31, wave = t >> 5;
  const int wr = wave >> 1, wc = wave & 1;            // 4x2 waves, wave tile 32x32
  const int bh = blockIdx.y, f0 = blockIdx.x * 128;
  const bool issuer = (wave == 0);
  const _Float16* kp = KpT + ((size_t)bh * 256 + f0) * 4096;
  const _Float16* vh = VhT + (size_t)bh * 64 * 4096;

  if (issuer) {
    tdm_load_2d(As[0], kp, 128, 64, 4096);
    tdm_load_2d(Bs[0], vh, 64, 64, 4096);
  }
  v8f acc[2][2] = {};
  for (int st = 0; st < 4096; st += 64) {
    const int cur = (st >> 6) & 1;
    if (st + 64 < 4096) {
      if (issuer) {
        tdm_load_2d(As[1 - cur], kp + st + 64, 128, 64, 4096);
        tdm_load_2d(Bs[1 - cur], vh + st + 64, 64, 64, 4096);
      }
      __builtin_amdgcn_s_wait_tensorcnt(2);
    } else {
      __builtin_amdgcn_s_wait_tensorcnt(0);
    }
    __syncthreads();
#pragma unroll
    for (int k2 = 0; k2 < 64; k2 += 32) {
      v16h af[2], bf[2];
#pragma unroll
      for (int i = 0; i < 2; ++i) af[i] = frag_a(As[cur], wr * 32 + i * 16, k2, lane, 64);
#pragma unroll
      for (int j = 0; j < 2; ++j) bf[j] = frag_b(Bs[cur], wc * 32 + j * 16, k2, lane, 64);
#pragma unroll
      for (int i = 0; i < 2; ++i)
#pragma unroll
        for (int j = 0; j < 2; ++j) acc[i][j] = wmma32(af[i], bf[j], acc[i][j]);
    }
    __syncthreads();
  }
#pragma unroll
  for (int i = 0; i < 2; ++i) {
    int fb = f0 + wr * 32 + i * 16 + ((lane >> 4) << 3);
#pragma unroll
    for (int j = 0; j < 2; ++j) {
      int d = wc * 32 + j * 16 + (lane & 15);
#pragma unroll
      for (int r = 0; r < 8; ++r)
        KvT[((size_t)bh * 64 + d) * 256 + fb + r] = (_Float16)acc[i][j][r];
    }
  }
}

// ----------- GEMM 4: num = q' @ kv, fused denominator + divide --------------
__global__ __launch_bounds__(256) void k_gemm_num(
    const _Float16* __restrict__ Qp, const _Float16* __restrict__ KvT,
    const float* __restrict__ Ksum, _Float16* __restrict__ Ao) {
  __shared__ __align__(16) _Float16 As[2][128 * 64];
  __shared__ __align__(16) _Float16 Bs[2][64 * 64];
  __shared__ float den[128];
  __shared__ float ks[256];
  const int t = threadIdx.x, lane = t & 31, wave = t >> 5;
  const int wr = wave >> 1, wc = wave & 1;
  const int bh = blockIdx.y, m0 = blockIdx.x * 128;
  const bool issuer = (wave == 0);
  const _Float16* qp = Qp + ((size_t)bh * 4096 + m0) * 256;
  const _Float16* kv = KvT + (size_t)bh * 64 * 256;
  const int sr = t >> 1, sc = (t & 1) * 32;   // 2 threads/row, 32 halves each

  if (issuer) {
    tdm_load_2d(As[0], qp, 128, 64, 256);
    tdm_load_2d(Bs[0], kv, 64, 64, 256);
  }
  ks[t] = Ksum[bh * 256 + t];
  if (t < 128) den[t] = 0.f;

  v8f acc[2][2] = {};
  for (int kt = 0; kt < 256; kt += 64) {
    const int cur = (kt >> 6) & 1;
    if (kt + 64 < 256) {
      if (issuer) {
        tdm_load_2d(As[1 - cur], qp + kt + 64, 128, 64, 256);
        tdm_load_2d(Bs[1 - cur], kv + kt + 64, 64, 64, 256);
      }
      __builtin_amdgcn_s_wait_tensorcnt(2);
    } else {
      __builtin_amdgcn_s_wait_tensorcnt(0);
    }
    __syncthreads();
    // denominator partial: read 32 A halves from LDS, dot with ksum (LDS)
    __align__(16) _Float16 h[32];
    cp16(h, &As[cur][sr * 64 + sc]);
    cp16(h + 16, &As[cur][sr * 64 + sc + 16]);
    float part = 0.f;
#pragma unroll
    for (int i = 0; i < 32; ++i) part += (float)h[i] * ks[kt + sc + i];
    atomicAdd(&den[sr], part);  // ds_add_f32
#pragma unroll
    for (int k2 = 0; k2 < 64; k2 += 32) {
      v16h af[2], bf[2];
#pragma unroll
      for (int i = 0; i < 2; ++i) af[i] = frag_a(As[cur], wr * 32 + i * 16, k2, lane, 64);
#pragma unroll
      for (int j = 0; j < 2; ++j) bf[j] = frag_b(Bs[cur], wc * 32 + j * 16, k2, lane, 64);
#pragma unroll
      for (int i = 0; i < 2; ++i)
#pragma unroll
        for (int j = 0; j < 2; ++j) acc[i][j] = wmma32(af[i], bf[j], acc[i][j]);
    }
    __syncthreads();
  }
  const int b = bh >> 4, hh = bh & 15;
#pragma unroll
  for (int i = 0; i < 2; ++i) {
    int ml = wr * 32 + i * 16 + ((lane >> 4) << 3);
#pragma unroll
    for (int j = 0; j < 2; ++j) {
      int d = wc * 32 + j * 16 + (lane & 15);
#pragma unroll
      for (int r = 0; r < 8; ++r) {
        float dn = den[ml + r] + 1e-6f;
        int s = m0 + ml + r;
        Ao[(size_t)(b * 4096 + s) * 1024 + hh * 64 + d] = (_Float16)(acc[i][j][r] / dn);
      }
    }
  }
}

// ----------------------- GEMM 5: output projection --------------------------
__global__ __launch_bounds__(256) void k_gemm_out(
    const _Float16* __restrict__ A, const _Float16* __restrict__ BwT,
    const float* __restrict__ bo, float* __restrict__ out) {
  __shared__ __align__(16) _Float16 As[2][128 * 64];
  __shared__ __align__(16) _Float16 Bs[2][128 * 64];
  const int t = threadIdx.x, lane = t & 31, wave = t >> 5;
  const int wr = wave >> 2, wc = wave & 3;
  const int m0 = blockIdx.y * 128, n0 = blockIdx.x * 128;
  const bool issuer = (wave == 0);
  const _Float16* pA = A + (size_t)m0 * 1024;
  const _Float16* pB = BwT + (size_t)n0 * 1024;

  if (issuer) {
    tdm_load_2d(As[0], pA, 128, 64, 1024);
    tdm_load_2d(Bs[0], pB, 128, 64, 1024);
  }
  v8f acc[4][2] = {};
  for (int kt = 0; kt < 1024; kt += 64) {
    const int cur = (kt >> 6) & 1;
    if (kt + 64 < 1024) {
      if (issuer) {
        tdm_load_2d(As[1 - cur], pA + kt + 64, 128, 64, 1024);
        tdm_load_2d(Bs[1 - cur], pB + kt + 64, 128, 64, 1024);
      }
      __builtin_amdgcn_s_wait_tensorcnt(2);
    } else {
      __builtin_amdgcn_s_wait_tensorcnt(0);
    }
    __syncthreads();
#pragma unroll
    for (int k2 = 0; k2 < 64; k2 += 32) {
      v16h af[4], bf[2];
#pragma unroll
      for (int i = 0; i < 4; ++i) af[i] = frag_a(As[cur], wr * 64 + i * 16, k2, lane, 64);
#pragma unroll
      for (int j = 0; j < 2; ++j) bf[j] = frag_b(Bs[cur], wc * 32 + j * 16, k2, lane, 64);
#pragma unroll
      for (int i = 0; i < 4; ++i)
#pragma unroll
        for (int j = 0; j < 2; ++j) acc[i][j] = wmma32(af[i], bf[j], acc[i][j]);
    }
    __syncthreads();
  }
#pragma unroll
  for (int i = 0; i < 4; ++i) {
    int mb = m0 + wr * 64 + i * 16 + ((lane >> 4) << 3);
#pragma unroll
    for (int j = 0; j < 2; ++j) {
      int n = n0 + wc * 32 + j * 16 + (lane & 15);
      float b = bo[n];
#pragma unroll
      for (int r = 0; r < 8; ++r)
        out[(size_t)(mb + r) * 1024 + n] = acc[i][j][r] + b;
    }
  }
}

// ---------------------------------------------------------------------------

extern "C" void kernel_launch(void* const* d_in, const int* in_sizes, int n_in,
                              void* d_out, int out_size, void* d_ws, size_t ws_size,
                              hipStream_t stream) {
  const float* x    = (const float*)d_in[0];
  const float* Wq   = (const float*)d_in[1];
  const float* bq   = (const float*)d_in[2];
  const float* Wk   = (const float*)d_in[3];
  const float* bk   = (const float*)d_in[4];
  const float* Wv   = (const float*)d_in[5];
  const float* bv   = (const float*)d_in[6];
  const float* rffW = (const float*)d_in[7];
  const float* rffb = (const float*)d_in[8];
  const float* Wo   = (const float*)d_in[9];
  const float* bo   = (const float*)d_in[10];
  float* out = (float*)d_out;

  char* ws = (char*)d_ws;
  size_t off = 0;
  auto carve = [&](size_t bytes) -> void* {
    void* p = ws + off;
    off = (off + bytes + 255) & ~(size_t)255;
    return p;
  };
  _Float16* xb     = (_Float16*)carve((size_t)16384 * 1024 * 2);
  _Float16* wqkvT  = (_Float16*)carve((size_t)3072 * 1024 * 2);
  _Float16* woT    = (_Float16*)carve((size_t)1024 * 1024 * 2);
  _Float16* rffwT  = (_Float16*)carve((size_t)256 * 64 * 2);
  _Float16* qh     = (_Float16*)carve((size_t)64 * 4096 * 64 * 2);
  _Float16* kh     = (_Float16*)carve((size_t)64 * 4096 * 64 * 2);
  _Float16* vhT    = (_Float16*)carve((size_t)64 * 64 * 4096 * 2);
  _Float16* qp     = (_Float16*)carve((size_t)64 * 4096 * 256 * 2);
  _Float16* kpT    = (_Float16*)carve((size_t)64 * 256 * 4096 * 2);
  _Float16* kvT    = (_Float16*)carve((size_t)64 * 64 * 256 * 2);
  float*    ksum   = (float*)carve((size_t)64 * 256 * 4);
  _Float16* ao     = (_Float16*)carve((size_t)16384 * 1024 * 2);

  // 1. fp32 -> fp16 conversion / transposed weight packing
  k_cvt_f16<<<(16384 * 1024) / 256, 256, 0, stream>>>(x, xb, 16384 * 1024);
  k_pack_wqkvT<<<(3072 * 1024) / 256, 256, 0, stream>>>(Wq, Wk, Wv, wqkvT);
  k_pack_woT<<<(1024 * 1024) / 256, 256, 0, stream>>>(Wo, woT);
  k_pack_rffwT<<<(256 * 64) / 256, 256, 0, stream>>>(rffW, rffwT);

  // 2. fused QKV projection (M=16384, N=3072, K=1024)
  k_gemm_qkv<<<dim3(24, 128), 256, 0, stream>>>(xb, wqkvT, bq, bk, bv, qh, kh, vhT);

  // 3. RFF feature maps (M=262144, N=256, K=64): q' row-major, k' transposed
  k_gemm_rff<<<dim3(2, 2048), 256, 0, stream>>>(qh, rffwT, rffb, qp, 0);
  k_gemm_rff<<<dim3(2, 2048), 256, 0, stream>>>(kh, rffwT, rffb, kpT, 1);

  // 4. k_sum over sequence
  k_ksum<<<64, 256, 0, stream>>>(kpT, ksum);

  // 5. kv = k'^T v per head (M=256, N=64, K=4096)
  k_gemm_kv<<<dim3(2, 64), 256, 0, stream>>>(kpT, vhT, kvT);

  // 6. num = q' @ kv, fused denominator + divide, merge heads
  k_gemm_num<<<dim3(32, 64), 256, 0, stream>>>(qp, kvT, ksum, ao);

  // 7. output projection (M=16384, N=1024, K=1024), fp32 out
  k_gemm_out<<<dim3(8, 128), 256, 0, stream>>>(ao, woT, bo, out);
}
